// GATNet_23295902613894
// MI455X (gfx1250) — matrix operand order repaired and checked by reading, not compiled, not run on hardware
//
#include <hip/hip_runtime.h>
#include <hip/hip_bf16.h>
#include <stdint.h>

// ---------------------------------------------------------------------------
// GATNet for MI455X (gfx1250): bf16 WMMA GEMMs (f32 accumulate) for all dense
// node-feature transforms + MLP; factorized GAT attention (per-node scores),
// L2-resident gather/scatter for edge aggregation; segment-max via int-bit
// atomicMax (exact on post-ReLU values).
// ---------------------------------------------------------------------------

typedef __bf16 bf16_t;
typedef __bf16 v16bf __attribute__((ext_vector_type(16)));
typedef float  v8f   __attribute__((ext_vector_type(8)));

#define HEADS     4
#define NN_NODES  30000
#define NE_EDGES  100000
#define NG_GRAPHS 256

union bfpack { bf16_t h[2]; unsigned u; };

static __device__ __forceinline__ unsigned pack2(float lo, float hi) {
  bfpack p;
  p.h[0] = (bf16_t)lo;
  p.h[1] = (bf16_t)hi;   // fuses to v_cvt_pk_bf16_f32
  return p.u;
}

// ===========================================================================
// WMMA GEMM:  C[M,N] = act( A[M,K] @ B[K,N] + bias )
// fp32 in/out, bf16 tiles in LDS, v_wmma_f32_16x16x32_bf16 core.
// Block = 256 thr (8 waves). Tile 128(M) x 128(N), BK = 32.
// Waves: 4 along M x 2 along N; each wave -> 32 x 64 (2x4 wmma tiles,
// 8 wmma / k-step; A frags reused over 4 B frags, B frags over 2 A frags).
//
// Out-of-range M rows / N cols are staged as *clamped* (in-bounds, garbage)
// values: they only feed output rows/cols the guarded epilogue never writes.
// Only the (at most one) partial-K step needs zero-fill -> scalar path.
// Alignment exploited: K even (float2 A loads), N multiple of 8 (float4 B).
// ===========================================================================
__global__ __launch_bounds__(256) void gemm_bf16_wmma(
    const float* __restrict__ A, const float* __restrict__ B,
    float* __restrict__ C, int M, int N, int K,
    const float* __restrict__ bias, int relu)
{
  __shared__ bf16_t As[128][40];   // [m][k], +8 pad
  __shared__ bf16_t Bs[128][40];   // [n][k] (transposed), +8 pad

  const int tid  = threadIdx.x;
  const int lane = tid & 31;
  const int wave = tid >> 5;
  const int half = lane >> 4;      // 0: lanes 0-15, 1: lanes 16-31
  const int lr   = lane & 15;
  const int wm   = wave & 3;       // wave tile row (M): 32 rows each
  const int wn   = wave >> 2;      // wave tile col (N): 64 cols each
  const int m0   = blockIdx.y * 128;
  const int n0   = blockIdx.x * 128;

  // staging coordinates
  const int arow = tid >> 1;            // 0..127
  const int akb  = (tid & 1) * 16;      // 0 or 16
  const int bk   = (tid >> 4) * 2;      // 0,2,..,30 (k pair)
  const int bnb  = (tid & 15) * 8;      // 0..120

  // clamped in-bounds sources for M/N edge tiles (garbage is harmless)
  const int rm  = min(m0 + arow, M - 1);
  const int ncl = min(n0 + bnb, N - 8);
  const float* __restrict__ arowp = A + (size_t)rm * K;

  v8f acc[2][4];
  #pragma unroll
  for (int i = 0; i < 2; ++i)
    #pragma unroll
    for (int t = 0; t < 4; ++t)
      #pragma unroll
      for (int r = 0; r < 8; ++r) acc[i][t][r] = 0.0f;

  for (int k0 = 0; k0 < K; k0 += 32) {
    __syncthreads();

    if (k0 + 32 <= K) {
      // ---- A tile 128x32 (fp32 -> bf16), 16 elems / thread, vector path ----
      const float* ap = arowp + k0 + akb;
      unsigned ua[8];
      #pragma unroll
      for (int j = 0; j < 8; ++j) {
        float2 f = *(const float2*)(ap + 2 * j);   // 8B-aligned (K even)
        ua[j] = pack2(f.x, f.y);
      }
      uint4* dst = (uint4*)(&As[arow][akb]);       // 16B-aligned
      dst[0] = make_uint4(ua[0], ua[1], ua[2], ua[3]);
      dst[1] = make_uint4(ua[4], ua[5], ua[6], ua[7]);

      // ---- B tile 32x128 -> Bs[n][k] transposed, pair-k packed stores ----
      const float* bp0 = B + (size_t)(k0 + bk) * N + ncl;   // 16B-aligned
      const float* bp1 = bp0 + N;
      float4 r0a = *(const float4*)(bp0);
      float4 r0b = *(const float4*)(bp0 + 4);
      float4 r1a = *(const float4*)(bp1);
      float4 r1b = *(const float4*)(bp1 + 4);
      unsigned ub[8];
      ub[0] = pack2(r0a.x, r1a.x); ub[1] = pack2(r0a.y, r1a.y);
      ub[2] = pack2(r0a.z, r1a.z); ub[3] = pack2(r0a.w, r1a.w);
      ub[4] = pack2(r0b.x, r1b.x); ub[5] = pack2(r0b.y, r1b.y);
      ub[6] = pack2(r0b.z, r1b.z); ub[7] = pack2(r0b.w, r1b.w);
      #pragma unroll
      for (int j = 0; j < 8; ++j)
        *(unsigned*)(&Bs[bnb + j][bk]) = ub[j];    // 4B-aligned (bk even)
    } else {
      // ---- partial-K step (at most once per block): zero-fill k >= K ----
      #pragma unroll
      for (int j = 0; j < 16; ++j) {
        int kk = k0 + akb + j;
        int rk = min(kk, K - 1);
        float v = arowp[rk];
        if (kk >= K) v = 0.0f;
        As[arow][akb + j] = (bf16_t)v;
      }
      int rk0 = min(k0 + bk, K - 1);
      int rk1 = min(k0 + bk + 1, K - 1);
      bool z0 = (k0 + bk >= K), z1 = (k0 + bk + 1 >= K);
      #pragma unroll
      for (int j = 0; j < 8; ++j) {
        float v0 = B[(size_t)rk0 * N + ncl + j];
        float v1 = B[(size_t)rk1 * N + ncl + j];
        if (z0) v0 = 0.0f;
        if (z1) v1 = 0.0f;
        *(unsigned*)(&Bs[bnb + j][bk]) = pack2(v0, v1);
      }
    }

    // prefetch next K tile into cache while this one computes
    if (k0 + 32 < K) {
      __builtin_prefetch(arowp + min(k0 + 32 + akb, K - 1), 0, 1);
      __builtin_prefetch(B + (size_t)min(k0 + 32 + bk, K - 1) * N + ncl, 0, 1);
    }

    __syncthreads();

    // ---- A fragments (16x32 bf16, ISA layout): pairs = 32b LDS reads ----
    union { v16bf v; unsigned u[8]; } af[2];
    #pragma unroll
    for (int i = 0; i < 2; ++i) {
      const unsigned* ar = (const unsigned*)(&As[wm * 32 + i * 16 + lr][0]);
      #pragma unroll
      for (int vv = 0; vv < 4; ++vv) af[i].u[vv] = ar[4 * half + vv];          // K = 8*half + 2v
      #pragma unroll
      for (int vv = 4; vv < 8; ++vv) af[i].u[vv] = ar[8 + 4 * half + vv - 4];  // K = 16 + ...
    }

    // ---- 4 B fragments (32x16 bf16) x 2 A fragments -> 8 WMMA ----
    #pragma unroll
    for (int t = 0; t < 4; ++t) {
      union { v16bf v; unsigned u[8]; } bfr;
      const unsigned* br = (const unsigned*)(&Bs[wn * 64 + t * 16 + lr][0]);
      #pragma unroll
      for (int vv = 0; vv < 4; ++vv) bfr.u[vv] = br[4 * half + vv];            // K = 8*half + 2v
      #pragma unroll
      for (int vv = 4; vv < 8; ++vv) bfr.u[vv] = br[8 + 4 * half + vv - 4];  // K = 16 + ...
      acc[0][t] = __builtin_amdgcn_wmma_f32_16x16x32_bf16(
          false, af[0].v, false, bfr.v, (short)0, acc[0][t], false, false);
      acc[1][t] = __builtin_amdgcn_wmma_f32_16x16x32_bf16(
          false, af[1].v, false, bfr.v, (short)0, acc[1][t], false, false);
    }
  }

  // ---- epilogue: C/D layout -> M = r + 8*half, N = lr ----
  #pragma unroll
  for (int t = 0; t < 4; ++t) {
    int gn = n0 + wn * 64 + t * 16 + lr;
    if (gn >= N) continue;
    float bv = bias ? bias[gn] : 0.0f;
    #pragma unroll
    for (int i = 0; i < 2; ++i) {
      #pragma unroll
      for (int r = 0; r < 8; ++r) {
        int gm = m0 + wm * 32 + i * 16 + 8 * half + r;
        if (gm < M) {
          float v = acc[i][t][r] + bv;
          if (relu) v = fmaxf(v, 0.0f);
          C[(size_t)gm * N + gn] = v;
        }
      }
    }
  }
}

// ===========================================================================
// Per-node attention scores: s[n,h,0] = sum_c xh[n,h,c]*att[h,c]
//                            s[n,h,1] = sum_c xh[n,h,c]*att[h,C+c]
// One wave32 per node, shuffle reduce.
// ===========================================================================
__global__ __launch_bounds__(256) void scores_kernel(
    const float* __restrict__ xh, const float* __restrict__ att,
    float* __restrict__ s, int Nn, int C)
{
  int gwave = (int)((blockIdx.x * (size_t)blockDim.x + threadIdx.x) >> 5);
  int lane  = threadIdx.x & 31;
  if (gwave >= Nn) return;
  const float* xn = xh + (size_t)gwave * HEADS * C;
  #pragma unroll
  for (int h = 0; h < HEADS; ++h) {
    const float* a0 = att + h * 2 * C;
    float s0 = 0.0f, s1 = 0.0f;
    for (int c = lane; c < C; c += 32) {
      float xv = xn[h * C + c];
      s0 += xv * a0[c];
      s1 += xv * a0[C + c];
    }
    #pragma unroll
    for (int o = 16; o > 0; o >>= 1) {
      s0 += __shfl_xor(s0, o, 32);
      s1 += __shfl_xor(s1, o, 32);
    }
    if (lane == 0) {
      s[(size_t)gwave * 2 * HEADS + 2 * h]     = s0;
      s[(size_t)gwave * 2 * HEADS + 2 * h + 1] = s1;
    }
  }
}

// ===========================================================================
// Per-edge: leaky_relu(si[dst]+sj[src]) then softmax over the 4 heads.
// ===========================================================================
__global__ __launch_bounds__(256) void alpha_kernel(
    const float* __restrict__ s, const int* __restrict__ ei,
    float* __restrict__ alpha, int E)
{
  int e = blockIdx.x * blockDim.x + threadIdx.x;
  if (e >= E) return;
  int src = ei[e];
  int dst = ei[E + e];
  float a[HEADS];
  float m = -3.0e38f;
  #pragma unroll
  for (int h = 0; h < HEADS; ++h) {
    float v = s[(size_t)dst * 2 * HEADS + 2 * h] +
              s[(size_t)src * 2 * HEADS + 2 * h + 1];
    v = (v > 0.0f) ? v : 0.2f * v;          // leaky_relu, slope 0.2
    a[h] = v;
    m = fmaxf(m, v);
  }
  float sum = 0.0f;
  #pragma unroll
  for (int h = 0; h < HEADS; ++h) { a[h] = __expf(a[h] - m); sum += a[h]; }
  float inv = 1.0f / sum;
  #pragma unroll
  for (int h = 0; h < HEADS; ++h) alpha[(size_t)e * HEADS + h] = a[h] * inv;
}

// ===========================================================================
// Scatter messages: agg[dst, f] += xh[src, f] * alpha[e, f/C]
// Consecutive threads -> consecutive f (coalesced gather + L2-local atomics).
// ===========================================================================
__global__ __launch_bounds__(256) void scatter_kernel(
    const float* __restrict__ xh, const float* __restrict__ alpha,
    const int* __restrict__ ei, float* __restrict__ agg,
    int E, int F, int C)
{
  long long idx = (long long)blockIdx.x * blockDim.x + threadIdx.x;
  long long total = (long long)E * F;
  if (idx >= total) return;
  int e = (int)(idx / F);
  int f = (int)(idx - (long long)e * F);
  int h = f / C;
  int src = ei[e];
  int dst = ei[E + e];
  float v = xh[(size_t)src * F + f] * alpha[(size_t)e * HEADS + h];
  atomicAdd(agg + (size_t)dst * F + f, v);
}

__global__ __launch_bounds__(256) void zero_kernel(float* __restrict__ p, long long n)
{
  long long i = (long long)blockIdx.x * blockDim.x + threadIdx.x;
  if (i < n) p[i] = 0.0f;
}

__global__ __launch_bounds__(256) void bias_relu_kernel(
    float* __restrict__ a, const float* __restrict__ b, long long n, int F)
{
  long long i = (long long)blockIdx.x * blockDim.x + threadIdx.x;
  if (i >= n) return;
  int f = (int)(i % F);
  a[i] = fmaxf(a[i] + b[f], 0.0f);
}

// segment_max: values are post-ReLU (>= 0) and gmax is 0-initialized, so
// int-bit atomicMax is exact.
__global__ __launch_bounds__(256) void segmax_kernel(
    const float* __restrict__ h, const int* __restrict__ batch,
    float* __restrict__ g, int Nn, int F)
{
  long long idx = (long long)blockIdx.x * blockDim.x + threadIdx.x;
  long long total = (long long)Nn * F;
  if (idx >= total) return;
  int n = (int)(idx / F);
  int f = (int)(idx - (long long)n * F);
  int gid = batch[n];
  atomicMax((int*)(g + (size_t)gid * F + f), __float_as_int(h[idx]));
}

// final [G,128] @ [128,1] + bo : one wave per graph
__global__ __launch_bounds__(256) void final_kernel(
    const float* __restrict__ g2, const float* __restrict__ Wo,
    const float* __restrict__ bo, float* __restrict__ out, int G)
{
  int gwave = (int)((blockIdx.x * (size_t)blockDim.x + threadIdx.x) >> 5);
  int lane  = threadIdx.x & 31;
  if (gwave >= G) return;
  float sum = 0.0f;
  #pragma unroll
  for (int c = lane; c < 128; c += 32)
    sum += g2[(size_t)gwave * 128 + c] * Wo[c];
  #pragma unroll
  for (int o = 16; o > 0; o >>= 1) sum += __shfl_xor(sum, o, 32);
  if (lane == 0) out[gwave] = sum + bo[0];
}

// ===========================================================================
// Host orchestration
// ===========================================================================
static void run_conv(const float* hin, int Kin, const float* W, const float* att,
                     const float* b, int C, float* xh, float* agg,
                     float* s, float* alpha, const int* ei, hipStream_t stream)
{
  const int F = HEADS * C;
  dim3 gg((F + 127) / 128, (NN_NODES + 127) / 128);
  gemm_bf16_wmma<<<gg, 256, 0, stream>>>(hin, W, xh, NN_NODES, F, Kin, nullptr, 0);
  scores_kernel<<<((size_t)NN_NODES * 32 + 255) / 256, 256, 0, stream>>>(xh, att, s, NN_NODES, C);
  alpha_kernel<<<(NE_EDGES + 255) / 256, 256, 0, stream>>>(s, ei, alpha, NE_EDGES);
  long long ntot = (long long)NN_NODES * F;
  zero_kernel<<<(unsigned)((ntot + 255) / 256), 256, 0, stream>>>(agg, ntot);
  long long etot = (long long)NE_EDGES * F;
  scatter_kernel<<<(unsigned)((etot + 255) / 256), 256, 0, stream>>>(xh, alpha, ei, agg, NE_EDGES, F, C);
  bias_relu_kernel<<<(unsigned)((ntot + 255) / 256), 256, 0, stream>>>(agg, b, ntot, F);
}

extern "C" void kernel_launch(void* const* d_in, const int* in_sizes, int n_in,
                              void* d_out, int out_size, void* d_ws, size_t ws_size,
                              hipStream_t stream) {
  const float* x    = (const float*)d_in[0];
  const int*   ei   = (const int*)d_in[1];
  const int*   batch= (const int*)d_in[2];
  const float* W1   = (const float*)d_in[3];
  const float* att1 = (const float*)d_in[4];
  const float* b1   = (const float*)d_in[5];
  const float* W2   = (const float*)d_in[6];
  const float* att2 = (const float*)d_in[7];
  const float* b2   = (const float*)d_in[8];
  const float* W3   = (const float*)d_in[9];
  const float* att3 = (const float*)d_in[10];
  const float* b3   = (const float*)d_in[11];
  const float* Wg1  = (const float*)d_in[12];
  const float* bg1  = (const float*)d_in[13];
  const float* Wg2  = (const float*)d_in[14];
  const float* bg2  = (const float*)d_in[15];
  const float* Wo   = (const float*)d_in[16];
  const float* bo   = (const float*)d_in[17];
  float* out = (float*)d_out;

  // workspace layout (floats)
  const size_t NF3 = (size_t)NN_NODES * 1248;     // biggest node-feature buffer
  float* P  = (float*)d_ws;                       // xh buffers (reused per layer)
  float* Q  = P + NF3;                            // agg / h buffers (reused)
  float* S  = Q + NF3;                            // node scores [N, H, 2]
  float* AL = S + (size_t)NN_NODES * 2 * HEADS;   // edge alphas [E, H]
  float* GM = AL + (size_t)NE_EDGES * HEADS;      // pooled [G, 1248]
  float* G1 = GM + (size_t)NG_GRAPHS * 1248;      // [G, 1024]
  float* G2 = G1 + (size_t)NG_GRAPHS * 1024;      // [G, 128]

  // conv1: 78 -> 4x78 (F=312), conv2: 312 -> 4x156 (F=624), conv3: 624 -> 4x312 (F=1248)
  run_conv(x, 78,  W1, att1, b1, 78,  P, Q, S, AL, ei, stream);
  run_conv(Q, 312, W2, att2, b2, 156, P, Q, S, AL, ei, stream);
  run_conv(Q, 624, W3, att3, b3, 312, P, Q, S, AL, ei, stream);

  // global max pool per graph (post-ReLU -> int-bit atomicMax exact, 0-init)
  long long gmn = (long long)NG_GRAPHS * 1248;
  zero_kernel<<<(unsigned)((gmn + 255) / 256), 256, 0, stream>>>(GM, gmn);
  long long hmn = (long long)NN_NODES * 1248;
  segmax_kernel<<<(unsigned)((hmn + 255) / 256), 256, 0, stream>>>(Q, batch, GM, NN_NODES, 1248);

  // MLP head
  {
    dim3 g1((1024 + 127) / 128, (NG_GRAPHS + 127) / 128);
    gemm_bf16_wmma<<<g1, 256, 0, stream>>>(GM, Wg1, G1, NG_GRAPHS, 1024, 1248, bg1, 1);
    dim3 g2((128 + 127) / 128, (NG_GRAPHS + 127) / 128);
    gemm_bf16_wmma<<<g2, 256, 0, stream>>>(G1, Wg2, G2, NG_GRAPHS, 128, 1024, bg2, 0);
    final_kernel<<<((size_t)NG_GRAPHS * 32 + 255) / 256, 256, 0, stream>>>(G2, Wo, bo, out, NG_GRAPHS);
  }
}
